// LatentODE_55430847922308
// MI455X (gfx1250) — matrix operand order repaired and checked by reading, not compiled, not run on hardware
//
#include <hip/hip_runtime.h>
#include <hip/hip_bf16.h>
#include <math.h>

typedef __attribute__((ext_vector_type(16))) _Float16 v16h;
typedef __attribute__((ext_vector_type(8)))  _Float16 v8h;
typedef __attribute__((ext_vector_type(8)))  float    v8f;

#define BB   256
#define TT   128
#define HH   64
#define LDIM 16
#define WW   128
#define SSN  5
#define ROWS 16      // batch rows per workgroup (== WMMA M)
#define NY   69      // S + H
#define YP   72      // padded f32 row pitch

__constant__ float c_scales[5] = {1.0e6f, 1.0e3f, 1.0e3f, 1.0e3f, 1.0e3f};

// ---------------- fast transcendentals (CDNA5 native TRANS32 ops) -----------

__device__ __forceinline__ float fast_tanh(float x) {
#if __has_builtin(__builtin_amdgcn_tanhf)
    return __builtin_amdgcn_tanhf(x);
#else
    float r;
    asm volatile("v_tanh_f32 %0, %1" : "=v"(r) : "v"(x));
    return r;
#endif
}
__device__ __forceinline__ float fast_sigmoid(float x) {
    // sigmoid(x) = 0.5*tanh(x/2) + 0.5  -> v_mul + v_tanh + v_fma
    return __builtin_fmaf(fast_tanh(0.5f * x), 0.5f, 0.5f);
}
__device__ __forceinline__ float fast_softplus(float x) {
    // native v_exp_f32 / v_log_f32 paths
    return x > 20.0f ? x : __logf(1.0f + __expf(x));
}

// ---------------- WMMA fragment helpers (CDNA5 16x16x32 f16) ----------------

// B operand (K x N), weights stored row-major (N, Kreal) in global f32.
// Layout: lane -> column n (lane&15), element e -> K = kbase + (e<8?e:e+8) + (lane>>4)*8.
__device__ __forceinline__ v16h load_bfrag_g(const float* w, int Kreal, int n,
                                             int kbase, int hs) {
    v16h v;
#pragma unroll
    for (int e = 0; e < 16; ++e) {
        int k = kbase + hs * 8 + (e < 8 ? e : e + 8);
        float x = (k < Kreal) ? w[n * Kreal + k] : 0.0f;
        v[e] = (_Float16)x;
    }
    return v;
}

// A operand (16 x 32 slice) from row-major f16 LDS activation buffer.
// Layout: lane -> row (lane&15); element e -> K = kbase + (e<8?e:e+8) + (lane>>4)*8.
// Both halves are contiguous 8-half (16B aligned) LDS loads.
__device__ __forceinline__ v16h load_afrag(const _Float16* src, int pitch,
                                           int lane, int kbase) {
    int r = lane & 15, hs = lane >> 4;
    union { v16h v; v8h h[2]; } u;
    const _Float16* p = src + r * pitch + kbase + hs * 8;
    u.h[0] = *(const v8h*)(p);
    u.h[1] = *(const v8h*)(p + 16);
    return u.v;
}

template <int KF>
__device__ __forceinline__ v8f wmma_tile(const _Float16* src, int spitch,
                                         const v16h* wf, float cinit, int lane) {
    v8f acc;
#pragma unroll
    for (int i = 0; i < 8; ++i) acc[i] = cinit;
#pragma unroll
    for (int f = 0; f < KF; ++f) {
        v16h a = load_afrag(src, spitch, lane, 32 * f);
        acc = __builtin_amdgcn_wmma_f32_16x16x32_f16(
            false, a, false, wf[f], (short)0, acc, false, false);
    }
    return acc;
}

// ---------------- shared memory layout ----------------

struct GruScratch {
    float hg[ROWS * 192];   // h @ whh^T
    float ig[ROWS * 192];   // x @ wih^T + b   (also reused as latent buf)
    float wih[192 * 6];
};

struct Smem {
    _Float16 xbuf[ROWS * 96];    // MLP layer-0 input, K padded 69->96
    _Float16 act1[ROWS * WW];
    _Float16 act2[ROWS * WW];
    _Float16 h16[ROWS * HH];     // GRU hidden, f16 mirror for WMMA
    float y[ROWS * YP];          // ODE state (f32 master)
    float yst[ROWS * YP];        // stage state / GRU f32 hidden
    float b0[WW], b1[WW], b2[WW], b3[HH];
    float whb[HH];  float bhb;
    float grub[192], grubn[HH];
    float bhl[LDIM], blh[HH];
    float beta[ROWS], dtv[ROWS];
    float xin[ROWS * 8];         // staged GRU inputs (ts, ys[0..4])
    union {
        float k[6][ROWS * YP];   // Tsit5 stage derivatives
        GruScratch g;
    } u;
};

// ---------------- kernel ----------------

__global__ __launch_bounds__(256)
void latentode_kernel(const float* __restrict__ ts, const float* __restrict__ ys,
                      const float* __restrict__ w0g, const float* __restrict__ b0g,
                      const float* __restrict__ w1g, const float* __restrict__ b1g,
                      const float* __restrict__ w2g, const float* __restrict__ b2g,
                      const float* __restrict__ w3g, const float* __restrict__ b3g,
                      const float* __restrict__ wihg, const float* __restrict__ whhg,
                      const float* __restrict__ grubg, const float* __restrict__ grubng,
                      const float* __restrict__ whlg, const float* __restrict__ bhlg,
                      const float* __restrict__ wlhg, const float* __restrict__ blhg,
                      const float* __restrict__ whbg, const float* __restrict__ bhbg,
                      float* __restrict__ out) {
    __shared__ Smem sm;

    const int tid  = threadIdx.x;
    const int lane = tid & 31;
    const int wid  = tid >> 5;      // wave id 0..7 (uniform per wave)
    const int ln   = lane & 15;
    const int hs   = lane >> 4;
    const int rb   = blockIdx.x * ROWS;

    // ---- persistent per-wave weight fragments in VGPRs ----
    v16h wf0[3], wf1[4], wf2[4], wf3[4], wg0[2], wg1[2];
#pragma unroll
    for (int f = 0; f < 3; ++f) wf0[f] = load_bfrag_g(w0g, 69, wid * 16 + ln, 32 * f, hs);
#pragma unroll
    for (int f = 0; f < 4; ++f) wf1[f] = load_bfrag_g(w1g, 128, wid * 16 + ln, 32 * f, hs);
#pragma unroll
    for (int f = 0; f < 4; ++f) wf2[f] = load_bfrag_g(w2g, 128, wid * 16 + ln, 32 * f, hs);
#pragma unroll
    for (int f = 0; f < 4; ++f) wf3[f] = load_bfrag_g(w3g, 128, (wid & 3) * 16 + ln, 32 * f, hs);
#pragma unroll
    for (int f = 0; f < 2; ++f) wg0[f] = load_bfrag_g(whhg, 64, wid * 16 + ln, 32 * f, hs);
#pragma unroll
    for (int f = 0; f < 2; ++f) wg1[f] = load_bfrag_g(whhg, 64, ((wid & 3) + 8) * 16 + ln, 32 * f, hs);

    // ---- stage small constants into LDS ----
    for (int i = tid; i < WW; i += 256) { sm.b0[i] = b0g[i]; sm.b1[i] = b1g[i]; sm.b2[i] = b2g[i]; }
    for (int i = tid; i < HH; i += 256) {
        sm.b3[i] = b3g[i]; sm.whb[i] = whbg[i]; sm.grubn[i] = grubng[i]; sm.blh[i] = blhg[i];
    }
    for (int i = tid; i < 192; i += 256) sm.grub[i] = grubg[i];
    for (int i = tid; i < 192 * 6; i += 256) sm.u.g.wih[i] = wihg[i];
    if (tid < LDIM) sm.bhl[tid] = bhlg[tid];
    if (tid == 0) sm.bhb = bhbg[0];
    for (int i = tid; i < ROWS * HH; i += 256) sm.h16[i] = (_Float16)0.0f;
    for (int i = tid; i < ROWS * YP; i += 256) sm.yst[i] = 0.0f;
    __syncthreads();

    // ================= GRU encoder (reversed time) =================
    for (int t = 0; t < TT; ++t) {
        const int tt = TT - 1 - t;
        if (tid < ROWS * 6) {
            int r = tid / 6, j = tid % 6;
            sm.xin[r * 8 + j] = (j == 0) ? ts[(rb + r) * TT + tt]
                                         : ys[((rb + r) * TT + tt) * SSN + (j - 1)];
        }
        __syncthreads();
        // ig = x @ wih^T + b  (scalar f32: ys magnitudes up to 1e6)
        for (int idx = tid; idx < ROWS * 192; idx += 256) {
            int r = idx / 192, n = idx % 192;
            float acc = sm.grub[n];
#pragma unroll
            for (int j = 0; j < 6; ++j) acc += sm.u.g.wih[n * 6 + j] * sm.xin[r * 8 + j];
            sm.u.g.ig[r * 192 + n] = acc;
        }
        // hg = h @ whh^T via WMMA (12 N-tiles over 8 waves)
        {
            v8f acc;
#pragma unroll
            for (int i = 0; i < 8; ++i) acc[i] = 0.0f;
#pragma unroll
            for (int f = 0; f < 2; ++f) {
                v16h a = load_afrag(sm.h16, HH, lane, 32 * f);
                acc = __builtin_amdgcn_wmma_f32_16x16x32_f16(
                    false, a, false, wg0[f], (short)0, acc, false, false);
            }
            int mb = hs * 8;
#pragma unroll
            for (int v = 0; v < 8; ++v) sm.u.g.hg[(mb + v) * 192 + wid * 16 + ln] = acc[v];
        }
        if (wid < 4) {  // wave-uniform; EXEC stays full inside
            v8f acc;
#pragma unroll
            for (int i = 0; i < 8; ++i) acc[i] = 0.0f;
#pragma unroll
            for (int f = 0; f < 2; ++f) {
                v16h a = load_afrag(sm.h16, HH, lane, 32 * f);
                acc = __builtin_amdgcn_wmma_f32_16x16x32_f16(
                    false, a, false, wg1[f], (short)0, acc, false, false);
            }
            int mb = hs * 8;
#pragma unroll
            for (int v = 0; v < 8; ++v) sm.u.g.hg[(mb + v) * 192 + (wid + 8) * 16 + ln] = acc[v];
        }
        __syncthreads();
        // gate combine + hidden update (native tanh path)
        for (int idx = tid; idx < ROWS * HH; idx += 256) {
            int r = idx / HH, j = idx % HH;
            float ir = sm.u.g.ig[r * 192 + j],        hr = sm.u.g.hg[r * 192 + j];
            float iz = sm.u.g.ig[r * 192 + 64 + j],   hz = sm.u.g.hg[r * 192 + 64 + j];
            float in_ = sm.u.g.ig[r * 192 + 128 + j], hn = sm.u.g.hg[r * 192 + 128 + j];
            float rg = fast_sigmoid(ir + hr);
            float zg = fast_sigmoid(iz + hz);
            float ng = fast_tanh(in_ + rg * (hn + sm.grubn[j]));
            float hp = sm.yst[r * YP + j];
            float hnew = ng + zg * (hp - ng);
            sm.yst[r * YP + j] = hnew;
            sm.h16[r * HH + j] = (_Float16)hnew;
        }
        __syncthreads();
    }

    // ================= latent decode: h0 = (h_enc Whl^T + bhl) Wlh^T + blh =====
    {
        int r = tid / 16, j = tid % 16;  // exactly 256 work items
        float acc = sm.bhl[j];
        for (int k = 0; k < HH; ++k) acc += whlg[j * HH + k] * sm.yst[r * YP + k];
        sm.u.g.ig[r * 16 + j] = acc;     // latent buffer (GRU scratch is free now)
    }
    __syncthreads();
    for (int idx = tid; idx < ROWS * HH; idx += 256) {
        int r = idx / HH, i = idx % HH;
        float acc = sm.blh[i];
#pragma unroll
        for (int k = 0; k < LDIM; ++k) acc += wlhg[i * LDIM + k] * sm.u.g.ig[r * 16 + k];
        sm.y[r * YP + SSN + i] = acc;
    }
    if (tid < ROWS * SSN) {
        int r = tid / SSN, s5 = tid % SSN;
        sm.y[r * YP + s5] = ys[((rb + r) * TT + 0) * SSN + s5] / c_scales[s5];
    }
    __syncthreads();

    const size_t outh_base = (size_t)BB * TT * SSN;
    // write t = 0
    if (tid < ROWS * SSN) {
        int r = tid / SSN, s5 = tid % SSN;
        out[((size_t)(rb + r) * TT + 0) * SSN + s5] = sm.y[r * YP + s5] * c_scales[s5];
    }
    for (int idx = tid; idx < ROWS * HH; idx += 256) {
        int r = idx / HH, j = idx % HH;
        out[outh_base + ((size_t)(rb + r) * TT + 0) * HH + j] = sm.y[r * YP + SSN + j];
    }

    // ================= ODE RHS (lambda; all barriers in uniform flow) =========
    auto rhs = [&](const float* src, float* kout) {
        __syncthreads();
        // build f16 MLP input x = [h, state*SCALES], pad 69->96; beta per row
        for (int idx = tid; idx < ROWS * 96; idx += 256) {
            int r = idx / 96, c = idx % 96;
            float v;
            if (c < HH)       v = src[r * YP + SSN + c];
            else if (c < NY)  v = src[r * YP + (c - HH)] * c_scales[c - HH];
            else              v = 0.0f;
            sm.xbuf[idx] = (_Float16)v;
        }
        if (tid < ROWS) {
            float acc = sm.bhb;
            for (int j = 0; j < HH; ++j) acc += sm.whb[j] * src[tid * YP + SSN + j];
            sm.beta[tid] = fast_sigmoid(acc);
        }
        __syncthreads();
        {   // layer 0: 69(->96) -> 128, softplus
            v8f acc = wmma_tile<3>(sm.xbuf, 96, wf0, sm.b0[wid * 16 + ln], lane);
            int mb = hs * 8;
#pragma unroll
            for (int v = 0; v < 8; ++v)
                sm.act1[(mb + v) * WW + wid * 16 + ln] = (_Float16)fast_softplus(acc[v]);
        }
        __syncthreads();
        {   // layer 1: 128 -> 128, softplus
            v8f acc = wmma_tile<4>(sm.act1, WW, wf1, sm.b1[wid * 16 + ln], lane);
            int mb = hs * 8;
#pragma unroll
            for (int v = 0; v < 8; ++v)
                sm.act2[(mb + v) * WW + wid * 16 + ln] = (_Float16)fast_softplus(acc[v]);
        }
        __syncthreads();
        {   // layer 2: 128 -> 128, softplus
            v8f acc = wmma_tile<4>(sm.act2, WW, wf2, sm.b2[wid * 16 + ln], lane);
            int mb = hs * 8;
#pragma unroll
            for (int v = 0; v < 8; ++v)
                sm.act1[(mb + v) * WW + wid * 16 + ln] = (_Float16)fast_softplus(acc[v]);
        }
        __syncthreads();
        if (wid < 4) {  // layer 3: 128 -> 64, dh = 0.1*tanh(1e-4*x)
            v8f acc = wmma_tile<4>(sm.act1, WW, wf3, sm.b3[wid * 16 + ln], lane);
            int mb = hs * 8;
#pragma unroll
            for (int v = 0; v < 8; ++v)
                kout[(mb + v) * YP + SSN + wid * 16 + ln] =
                    0.1f * fast_tanh(1.0e-4f * acc[v]);
        } else {        // waves 4..7: SEIR dstate concurrently
            int t2 = tid - 128;
            if (t2 < ROWS * SSN) {
                int r = t2 / SSN, s5 = t2 % SSN;
                float Sc = src[r * YP + 0] * 1.0e6f;
                float E  = src[r * YP + 1] * 1.0e3f;
                float I  = src[r * YP + 2] * 1.0e3f;
                float Aa = src[r * YP + 3] * 1.0e3f;
                const float kk = 0.526f, aa = 0.244f, ii = 0.244f, p = 0.667f, ff = 0.98f;
                float Lsum = 0.5f * I + Aa;           // ee*E + (1-q)*I + dd*A
                float bSL = sm.beta[r] * Sc * Lsum;
                float d;
                if      (s5 == 0) d = -bSL;
                else if (s5 == 1) d = bSL - kk * E;
                else if (s5 == 2) d = p * kk * E - aa * I;
                else if (s5 == 3) d = (1.0f - p) * kk * E - ii * Aa;
                else              d = ff * aa * I + ii * Aa;
                kout[r * YP + s5] = d / c_scales[s5];
            }
        }
        __syncthreads();
    };

    auto stage_comb = [&](int nc, const float* coef) {
        for (int idx = tid; idx < ROWS * NY; idx += 256) {
            int r = idx / NY, c = idx % NY;
            float s = 0.0f;
            for (int j = 0; j < nc; ++j) s += coef[j] * sm.u.k[j][r * YP + c];
            sm.yst[r * YP + c] = sm.y[r * YP + c] + sm.dtv[r] * s;
        }
        __syncthreads();
    };

    const float C2[1] = {0.161f};
    const float C3[2] = {-0.008480655492356989f, 0.335480655492357f};
    const float C4[3] = {2.8971530571054935f, -6.359448489975075f, 4.3622954328695815f};
    const float C5[4] = {5.325864828439257f, -11.748883564062828f, 7.4955393428898365f,
                         -0.09249506636175525f};
    const float C6[5] = {5.86145544294642f, -12.92096931784711f, 8.159367898576159f,
                         -0.071584973281401f, -0.028269050394068383f};
    const float CB[6] = {0.09646076681806523f, 0.01f, 0.4798896504144996f,
                         1.379008574103742f, -3.290069515436081f, 2.324710524099774f};

    // ================= ODE time loop =================
    for (int iv = 0; iv < TT - 1; ++iv) {
        if (tid < ROWS)
            sm.dtv[tid] = (ts[(rb + tid) * TT + iv + 1] - ts[(rb + tid) * TT + iv]) * 0.5f;
        __syncthreads();
#pragma unroll 1
        for (int sub = 0; sub < 2; ++sub) {
            rhs(sm.y, sm.u.k[0]);
            stage_comb(1, C2); rhs(sm.yst, sm.u.k[1]);
            stage_comb(2, C3); rhs(sm.yst, sm.u.k[2]);
            stage_comb(3, C4); rhs(sm.yst, sm.u.k[3]);
            stage_comb(4, C5); rhs(sm.yst, sm.u.k[4]);
            stage_comb(5, C6); rhs(sm.yst, sm.u.k[5]);
            for (int idx = tid; idx < ROWS * NY; idx += 256) {
                int r = idx / NY, c = idx % NY;
                float s = 0.0f;
#pragma unroll
                for (int j = 0; j < 6; ++j) s += CB[j] * sm.u.k[j][r * YP + c];
                sm.y[r * YP + c] += sm.dtv[r] * s;
            }
            __syncthreads();
        }
        // write t = iv+1
        if (tid < ROWS * SSN) {
            int r = tid / SSN, s5 = tid % SSN;
            out[((size_t)(rb + r) * TT + iv + 1) * SSN + s5] = sm.y[r * YP + s5] * c_scales[s5];
        }
        for (int idx = tid; idx < ROWS * HH; idx += 256) {
            int r = idx / HH, j = idx % HH;
            out[outh_base + ((size_t)(rb + r) * TT + iv + 1) * HH + j] = sm.y[r * YP + SSN + j];
        }
        __syncthreads();
    }
}

extern "C" void kernel_launch(void* const* d_in, const int* in_sizes, int n_in,
                              void* d_out, int out_size, void* d_ws, size_t ws_size,
                              hipStream_t stream) {
    (void)in_sizes; (void)n_in; (void)out_size; (void)d_ws; (void)ws_size;
    const float* ts    = (const float*)d_in[0];
    const float* ys    = (const float*)d_in[1];
    const float* w0    = (const float*)d_in[2];
    const float* b0    = (const float*)d_in[3];
    const float* w1    = (const float*)d_in[4];
    const float* b1    = (const float*)d_in[5];
    const float* w2    = (const float*)d_in[6];
    const float* b2    = (const float*)d_in[7];
    const float* w3    = (const float*)d_in[8];
    const float* b3    = (const float*)d_in[9];
    const float* wih   = (const float*)d_in[10];
    const float* whh   = (const float*)d_in[11];
    const float* grub  = (const float*)d_in[12];
    const float* grubn = (const float*)d_in[13];
    const float* whl   = (const float*)d_in[14];
    const float* bhl   = (const float*)d_in[15];
    const float* wlh   = (const float*)d_in[16];
    const float* blh   = (const float*)d_in[17];
    const float* whb   = (const float*)d_in[18];
    const float* bhb   = (const float*)d_in[19];

    latentode_kernel<<<dim3(BB / ROWS), dim3(256), 0, stream>>>(
        ts, ys, w0, b0, w1, b1, w2, b2, w3, b3,
        wih, whh, grub, grubn, whl, bhl, wlh, blh, whb, bhb,
        (float*)d_out);
}
